// MambaVisionMixer_31396210934375
// MI455X (gfx1250) — matrix operand order repaired and checked by reading, not compiled, not run on hardware
//
#include <hip/hip_runtime.h>
#include <hip/math_functions.h>

// ---------------------------------------------------------------------------
// Mamba mixer forward for MI455X (gfx1250, wave32).
//  * all GEMMs: v_wmma_f32_16x16x32_f16, f16 operands, f32 accumulation
//  * A-tiles staged to LDS with global_load_async_to_lds_b128 (ASYNCcnt)
//    and consumed with ds_load_b128
//  * operands pre-converted to f16 so fragment loads are 128-bit vectors
// ---------------------------------------------------------------------------

typedef __attribute__((ext_vector_type(16))) _Float16 v16h;
typedef __attribute__((ext_vector_type(8)))  _Float16 v8h;
typedef __attribute__((ext_vector_type(8)))  float    v8f;

constexpr int LSEQ   = 4096;
constexpr int NBATCH = 2;
constexpr int DMODEL = 512;
constexpr int DINNER = 1024;
constexpr int DHALF  = 512;
constexpr int DSTATE = 16;
constexpr int DTRANK = 32;

__device__ __forceinline__ float silu_f(float x) {
    return x / (1.0f + __expf(-x));
}
__device__ __forceinline__ float softplus_f(float x) {
    return (x > 20.0f) ? x : __logf(1.0f + __expf(x));
}

// ---------------------------------------------------------------------------
// f32 -> f16 conversion (weights / hidden, one time)
// ---------------------------------------------------------------------------
__global__ __launch_bounds__(256)
void cvt_f32_to_f16(const float* __restrict__ src, _Float16* __restrict__ dst, int n)
{
    const int i = blockIdx.x * 256 + threadIdx.x;
    if (i < n) dst[i] = (_Float16)src[i];
}

// ---------------------------------------------------------------------------
// Async-staged WMMA GEMM:  C[M,N] = A[M,K] * W[N,K]^T
//   A: f16, row-major with leading dimension lda (lda >= KT)
//   W: f16, row-major [N, KT]
//   C_MODE 0: f32 row-major [M,N]
//   C_MODE 1: f32 channel-major [B, c_ch, LSEQ]  (m = b*LSEQ + l)
//   C_MODE 2: f16 row-major [M,N]
//   EPI    1: softplus(val + bias[n])
//
// Block = 256 threads = 8 waves. blockIdx.y selects the 16-row M tile which
// is staged once into LDS via async DMA and shared by all waves; each wave
// owns one 16-col N tile (tn = (blockIdx.x*8 + wave)*16).
// ---------------------------------------------------------------------------
template<int C_MODE, int EPI, int KT>
__global__ __launch_bounds__(256)
void wmma_gemm_async(const _Float16* __restrict__ A, const _Float16* __restrict__ W,
                     void* __restrict__ Cout, const float* __restrict__ bias,
                     int M, int N, int lda, int c_ch)
{
    __shared__ _Float16 sA[16 * KT];

    const int lane = threadIdx.x & 31;
    const int wave = threadIdx.x >> 5;
    const int tm   = blockIdx.y << 4;
    const int tn   = (blockIdx.x * 8 + wave) << 4;

    // ---- stage A tile (16 x KT f16 = 32*KT bytes) into LDS via async DMA ----
    {
        const unsigned lds_base = (unsigned)(uintptr_t)(&sA[0]);
        for (int c = threadIdx.x; c < 2 * KT; c += 256) {   // 16-byte chunks
            const int e   = c * 8;            // f16 element index in tile
            const int row = e / KT;           // KT is a power of two -> shifts
            const int col = e % KT;
            const _Float16* g = A + (size_t)(tm + row) * lda + col;
            const unsigned loff = lds_base + (unsigned)(e * 2);
            asm volatile("global_load_async_to_lds_b128 %0, %1, off"
                         :: "v"(loff), "v"(g) : "memory");
        }
        asm volatile("s_wait_asynccnt 0x0" ::: "memory");
    }
    __syncthreads();

    const int half = lane >> 4;   // lane group selects K sub-block
    const int r    = lane & 15;   // row within tile

    if (tn < N) {
        const _Float16* __restrict__ wrow = W + (size_t)(tn + r) * KT;
        v8f acc = {};

        for (int k0 = 0; k0 < KT; k0 += 32) {
            if (k0 + 32 < KT) __builtin_prefetch(wrow + k0 + 32, 0, 3);

            // A fragment: VGPR 0-3 hold K = k0+16*half .. +7 (one b128 from LDS),
            //             VGPR 4-7 hold K = k0+16*half+8 .. +7
            const v8h a0 = *(const v8h*)(sA + r * KT + k0 + 16 * half);
            const v8h a1 = *(const v8h*)(sA + r * KT + k0 + 16 * half + 8);
            const v16h af = __builtin_shufflevector(a0, a1,
                0, 1, 2, 3, 4, 5, 6, 7, 8, 9, 10, 11, 12, 13, 14, 15);

            // B fragment: lane n row, K = k0+16*half .. +15 contiguous (2 x b128)
            const v8h b0 = *(const v8h*)(wrow + k0 + 16 * half);
            const v8h b1 = *(const v8h*)(wrow + k0 + 16 * half + 8);
            const v16h bf = __builtin_shufflevector(b0, b1,
                0, 1, 2, 3, 4, 5, 6, 7, 8, 9, 10, 11, 12, 13, 14, 15);

            acc = __builtin_amdgcn_wmma_f32_16x16x32_f16(
                false, af, false, bf, (short)0, acc, false, false);
        }

        // C/D fragment: VGPR v -> row = tm + v + 8*half, col = tn + (lane&15)
        const int col = tn + r;
#pragma unroll
        for (int v = 0; v < 8; ++v) {
            const int row = tm + v + 8 * half;
            float val = acc[v];
            if (EPI == 1) val = softplus_f(val + bias[col]);
            if (C_MODE == 0) {
                ((float*)Cout)[(size_t)row * N + col] = val;
            } else if (C_MODE == 1) {
                ((float*)Cout)[(size_t)(row / LSEQ) * c_ch * LSEQ
                               + (size_t)col * LSEQ + (row % LSEQ)] = val;
            } else {
                ((_Float16*)Cout)[(size_t)row * N + col] = (_Float16)val;
            }
        }
    }
}

// ---------------------------------------------------------------------------
// Depthwise conv1d (k=4, SAME: pad_lo=1, pad_hi=2) + SiLU.
//   x -> xc   (f32, [B, DHALF, L], for the scan)
//     -> xch16(f16, row-major [B*L, DHALF], for x_proj GEMM)
//   z -> yzh  (f16, row-major [B*L, DINNER], cols 512..1023, for out_proj)
// ---------------------------------------------------------------------------
__global__ __launch_bounds__(256)
void conv_silu_kernel(const float* __restrict__ xz,
                      const float* __restrict__ wx,
                      const float* __restrict__ wz,
                      float* __restrict__ xc,
                      _Float16* __restrict__ xch16,
                      _Float16* __restrict__ yzh)
{
    const int idx = blockIdx.x * blockDim.x + threadIdx.x;
    const int total = NBATCH * DHALF * LSEQ;
    if (idx >= total) return;

    const int l = idx % LSEQ;
    const int d = (idx / LSEQ) % DHALF;
    const int b = idx / (LSEQ * DHALF);

    const float* xrow = xz + ((size_t)b * DINNER + d) * LSEQ;
    const float* zrow = xz + ((size_t)b * DINNER + DHALF + d) * LSEQ;

    float ax = 0.0f, az = 0.0f;
#pragma unroll
    for (int j = 0; j < 4; ++j) {
        const int li = l - 1 + j;             // SAME padding for k=4
        if (li >= 0 && li < LSEQ) {
            ax = fmaf(xrow[li], wx[d * 4 + j], ax);
            az = fmaf(zrow[li], wz[d * 4 + j], az);
        }
    }
    const float sx = silu_f(ax);
    const float sz = silu_f(az);
    const size_t m = (size_t)b * LSEQ + l;
    xc[((size_t)b * DHALF + d) * LSEQ + l] = sx;
    xch16[m * DHALF + d]                   = (_Float16)sx;
    yzh[m * DINNER + DHALF + d]            = (_Float16)sz;
}

// ---------------------------------------------------------------------------
// Selective scan: one thread per (b,d) channel, 16 states in registers.
// B/C vectors come from the f16 row-major xdblh (32 contiguous halves per l).
// y written as f16 into yzh cols 0..511 (out_proj A operand).
// ---------------------------------------------------------------------------
__global__ __launch_bounds__(128)
void scan_kernel(const float* __restrict__ xc,
                 const _Float16* __restrict__ xdblh,
                 const float* __restrict__ delta,
                 const float* __restrict__ A_log,
                 const float* __restrict__ Dvec,
                 _Float16* __restrict__ yzh)
{
    const int idx = blockIdx.x * blockDim.x + threadIdx.x;
    if (idx >= NBATCH * DHALF) return;
    const int b = idx / DHALF;
    const int d = idx % DHALF;

    float Ad[DSTATE];
#pragma unroll
    for (int n = 0; n < DSTATE; ++n)
        Ad[n] = -__expf(A_log[d * DSTATE + n]);
    const float Dv = Dvec[d];

    const float* xp = xc    + ((size_t)b * DHALF + d) * LSEQ;
    const float* dp = delta + ((size_t)b * DHALF + d) * LSEQ;

    float h[DSTATE];
#pragma unroll
    for (int n = 0; n < DSTATE; ++n) h[n] = 0.0f;

    for (int l = 0; l < LSEQ; ++l) {
        const size_t m = (size_t)b * LSEQ + l;
        const float dv = dp[l];
        const float xv = xp[l];
        const float dx = dv * xv;
        float y = Dv * xv;
        const _Float16* bc = xdblh + m * 64 + DTRANK;   // B[0..15], C[0..15]
#pragma unroll
        for (int n = 0; n < DSTATE; ++n) {
            const float dA = __expf(dv * Ad[n]);
            h[n] = fmaf(dA, h[n], dx * (float)bc[n]);
            y    = fmaf(h[n], (float)bc[DSTATE + n], y);
        }
        yzh[m * DINNER + d] = (_Float16)y;
    }
}

// ---------------------------------------------------------------------------
// Launcher. Workspace layout:
//   f32: xz[B,1024,L] | xc[B,512,L] | delta[B,512,L]
//   f16: hiddenh[8192,512] | xch16[8192,512] | xdblh[8192,64] | yzh[8192,1024]
//        wih[1024,512] | wxh[64,512] | wdh[512,32] | woh[512,1024]
// ---------------------------------------------------------------------------
extern "C" void kernel_launch(void* const* d_in, const int* in_sizes, int n_in,
                              void* d_out, int out_size, void* d_ws, size_t ws_size,
                              hipStream_t stream)
{
    const float* hidden     = (const float*)d_in[0];
    const float* in_proj_w  = (const float*)d_in[1];
    const float* conv_x_w   = (const float*)d_in[2];
    const float* conv_z_w   = (const float*)d_in[3];
    const float* x_proj_w   = (const float*)d_in[4];
    const float* dt_proj_w  = (const float*)d_in[5];
    const float* dt_proj_b  = (const float*)d_in[6];
    const float* A_log      = (const float*)d_in[7];
    const float* Dvec       = (const float*)d_in[8];
    const float* out_proj_w = (const float*)d_in[9];
    float* out = (float*)d_out;

    const int M = NBATCH * LSEQ;   // 8192

    // ---- carve workspace (all offsets 16-float aligned) ----
    float* wsf = (float*)d_ws;
    size_t off = 0;
    float* xz    = wsf + off; off += (size_t)NBATCH * DINNER * LSEQ;  // 8.4M f
    float* xc    = wsf + off; off += (size_t)NBATCH * DHALF  * LSEQ;  // 4.2M f
    float* delta = wsf + off; off += (size_t)NBATCH * DHALF  * LSEQ;  // 4.2M f

    _Float16* wsh = (_Float16*)(wsf + off);
    size_t hoff = 0;
    _Float16* hiddenh = wsh + hoff; hoff += (size_t)M * DMODEL;   // 4.2M h
    _Float16* xch16   = wsh + hoff; hoff += (size_t)M * DHALF;    // 4.2M h
    _Float16* xdblh   = wsh + hoff; hoff += (size_t)M * 64;       // 0.5M h
    _Float16* yzh     = wsh + hoff; hoff += (size_t)M * DINNER;   // 8.4M h
    _Float16* wih     = wsh + hoff; hoff += (size_t)DINNER * DMODEL;
    _Float16* wxh     = wsh + hoff; hoff += (size_t)64 * DHALF;
    _Float16* wdh     = wsh + hoff; hoff += (size_t)DHALF * DTRANK;
    _Float16* woh     = wsh + hoff; hoff += (size_t)DMODEL * DINNER;

    // ---- 0) convert GEMM operands to f16 ----
    {
        int n;
        n = M * DMODEL;
        cvt_f32_to_f16<<<(n + 255) / 256, 256, 0, stream>>>(hidden, hiddenh, n);
        n = DINNER * DMODEL;
        cvt_f32_to_f16<<<(n + 255) / 256, 256, 0, stream>>>(in_proj_w, wih, n);
        n = 64 * DHALF;
        cvt_f32_to_f16<<<(n + 255) / 256, 256, 0, stream>>>(x_proj_w, wxh, n);
        n = DHALF * DTRANK;
        cvt_f32_to_f16<<<(n + 255) / 256, 256, 0, stream>>>(dt_proj_w, wdh, n);
        n = DMODEL * DINNER;
        cvt_f32_to_f16<<<(n + 255) / 256, 256, 0, stream>>>(out_proj_w, woh, n);
    }

    // ---- 1) in_proj: xz[b,e,l] (f32 cm), M=8192 N=1024 K=512 ----
    {
        const int N = DINNER;
        dim3 grid((N / 16 + 7) / 8, M / 16);
        wmma_gemm_async<1, 0, DMODEL><<<grid, 256, 0, stream>>>(
            hiddenh, wih, xz, nullptr, M, N, DMODEL, DINNER);
    }

    // ---- 2) depthwise conv + SiLU ----
    {
        const int total = NBATCH * DHALF * LSEQ;
        conv_silu_kernel<<<(total + 255) / 256, 256, 0, stream>>>(
            xz, conv_x_w, conv_z_w, xc, xch16, yzh);
    }

    // ---- 3) x_proj: xdblh (f16 rm [m,64]), N=64 K=512 ----
    {
        const int N = 64;
        dim3 grid((N / 16 + 7) / 8, M / 16);
        wmma_gemm_async<2, 0, DHALF><<<grid, 256, 0, stream>>>(
            xch16, wxh, xdblh, nullptr, M, N, DHALF, 0);
    }

    // ---- 4) dt_proj + softplus: delta (f32 cm), N=512 K=32, lda=64 ----
    {
        const int N = DHALF;
        dim3 grid((N / 16 + 7) / 8, M / 16);
        wmma_gemm_async<1, 1, DTRANK><<<grid, 256, 0, stream>>>(
            xdblh, wdh, delta, dt_proj_b, M, N, 64, DHALF);
    }

    // ---- 5) selective scan -> yzh cols 0..511 ----
    {
        const int total = NBATCH * DHALF;
        scan_kernel<<<(total + 127) / 128, 128, 0, stream>>>(
            xc, xdblh, delta, A_log, Dvec, yzh);
    }

    // ---- 6) out_proj: out[b,l,e] (f32 rm), N=512 K=1024 ----
    {
        const int N = DMODEL;
        dim3 grid((N / 16 + 7) / 8, M / 16);
        wmma_gemm_async<0, 0, DINNER><<<grid, 256, 0, stream>>>(
            yzh, woh, out, nullptr, M, N, DINNER, 0);
    }
}
